// _NonLocalBlockND_pair_83829171683430
// MI455X (gfx1250) — matrix operand order repaired and checked by reading, not compiled, not run on hardware
//
#include <hip/hip_runtime.h>

// ---------------------------------------------------------------------------
// NonLocalBlockND pair for MI455X (gfx1250, wave32, WMMA bf16, async->LDS)
// ---------------------------------------------------------------------------
#define Bn 4
#define C_  256
#define CI_ 128
#define N_  4096      // H*W = 64*64
#define KB_ 32        // keys per attention step
#define EPSV 1e-5f

typedef __attribute__((ext_vector_type(16))) __bf16 v16bf;
typedef __attribute__((ext_vector_type(8)))  __bf16 v8bf;
typedef __attribute__((ext_vector_type(8)))  float  v8f;

__device__ inline __bf16 f2bf(float f) {
  union { float f; unsigned u; } v; v.f = f;
  unsigned r = v.u + 0x7FFFu + ((v.u >> 16) & 1u);   // RNE
  unsigned short h = (unsigned short)(r >> 16);
  return __builtin_bit_cast(__bf16, h);
}

__device__ inline v8f wmma_bf16(v16bf a, v16bf b, v8f c) {
  return __builtin_amdgcn_wmma_f32_16x16x32_bf16(
      /*neg_a=*/false, a, /*neg_b=*/false, b,
      /*c_mod=*/(short)0, c, /*reuse_a=*/false, /*reuse_b=*/false);
}

// Low 32 bits of the generic address of a __shared__ object = LDS byte address
// (flat aperture mapping truncates to addr[31:0] for LDS).
__device__ inline unsigned lds_off(const void* p) {
  return (unsigned)(uintptr_t)p;
}

// One wave-wide async DMA: each lane moves 16 bytes global->LDS (ASYNCcnt).
__device__ inline void async_b128(unsigned lds_byte, unsigned gbyte,
                                  const void* base) {
  asm volatile("global_load_async_to_lds_b128 %0, %1, %2"
               :: "v"(lds_byte), "v"(gbyte),
                  "s"((unsigned long long)(uintptr_t)base)
               : "memory");
}
__device__ inline void wait_async0() {
  asm volatile("s_wait_asynccnt 0x0" ::: "memory");
}

// A-matrix (16x32 bf16, M x K) tile:
//   lane = 16*g + m (m = row). halves 0..7 : K = kk + 16g + (0..7)
//                              halves 8..15: K = kk + 16g + 8 + (0..7)
__device__ inline v16bf load_Atile(const __bf16* __restrict__ row, int kk, int g) {
  v8bf lo = *(const v8bf*)(row + kk + 16 * g);
  v8bf hi = *(const v8bf*)(row + kk + 16 * g + 8);
  v16bf a;
#pragma unroll
  for (int i = 0; i < 8; ++i) { a[i] = lo[i]; a[8 + i] = hi[i]; }
  return a;
}
__device__ inline v16bf load_Atile_lds(const __bf16* row, int g) {
  v8bf lo = *(const v8bf*)(row + 16 * g);
  v8bf hi = *(const v8bf*)(row + 16 * g + 8);
  v16bf a;
#pragma unroll
  for (int i = 0; i < 8; ++i) { a[i] = lo[i]; a[8 + i] = hi[i]; }
  return a;
}

// ---------------------------------------------------------------------------
// 1) (B,C,N) f32  ->  token-major (B*N, C) bf16, tiled transpose
// ---------------------------------------------------------------------------
__global__ void __launch_bounds__(256)
k_pack_tokens(const float* __restrict__ in, __bf16* __restrict__ out) {
  __shared__ float tile[32][33];
  const int b  = blockIdx.z;
  const int n0 = blockIdx.x * 32;
  const int c0 = blockIdx.y * 32;
#pragma unroll
  for (int i = threadIdx.y; i < 32; i += 8)
    tile[i][threadIdx.x] = in[((size_t)(b * C_ + c0 + i)) * N_ + n0 + threadIdx.x];
  __syncthreads();
#pragma unroll
  for (int i = threadIdx.y; i < 32; i += 8)
    out[((size_t)b * N_ + n0 + i) * C_ + c0 + threadIdx.x] = f2bf(tile[threadIdx.x][i]);
}

__global__ void __launch_bounds__(256)
k_cvt(const float* __restrict__ in, __bf16* __restrict__ out, int n) {
  int i = blockIdx.x * 256 + threadIdx.x;
  if (i < n) out[i] = f2bf(in[i]);
}

// ---------------------------------------------------------------------------
// 2) Projection GEMM: out[n][o] = sum_c X[n][c] * W[o][c] + bias[o]
//    Weight matrix (64 KB) staged once per block into LDS via async DMA.
// ---------------------------------------------------------------------------
__global__ void __launch_bounds__(128)
k_proj(const __bf16* __restrict__ X, const __bf16* __restrict__ Wm,
       const float* __restrict__ bias, __bf16* __restrict__ out, int store_mode) {
  __shared__ __bf16 wS[CI_ * C_];                  // 64 KB
  const int tid  = threadIdx.x;
  const int lane = tid & 31;
  const int wave = tid >> 5;
  const int g = lane >> 4, m = lane & 15;
  const int token0 = (blockIdx.x * 4 + wave) * 16;

  {                                                // 512 B per thread-lane
    unsigned l0 = lds_off(wS) + (unsigned)tid * 512u;
    unsigned g0 = (unsigned)tid * 512u;
#pragma unroll
    for (int i = 0; i < 32; ++i)
      async_b128(l0 + i * 16, g0 + i * 16, Wm);
  }
  wait_async0();
  __syncthreads();

  v8f acc[8];
#pragma unroll
  for (int t = 0; t < 8; ++t) acc[t] = (v8f){0.f,0.f,0.f,0.f,0.f,0.f,0.f,0.f};

  const __bf16* arow = X + (size_t)(token0 + m) * C_;
#pragma unroll 2
  for (int kk = 0; kk < C_; kk += 32) {
    v16bf A = load_Atile(arow, kk, g);
#pragma unroll
    for (int ot = 0; ot < 8; ++ot) {
      // B tile: lane = 16g + col(o); halves j: K = kk + 16g + j (contiguous)
      v16bf Bt = *(const v16bf*)(wS + (size_t)(ot * 16 + m) * C_ + kk + 16 * g);
      acc[ot] = wmma_bf16(A, Bt, acc[ot]);
    }
  }
#pragma unroll
  for (int ot = 0; ot < 8; ++ot) {
    const int o = ot * 16 + m;
    const float bsum = bias[o];
#pragma unroll
    for (int r = 0; r < 8; ++r) {
      const int token = token0 + r + 8 * g;
      __bf16 h = f2bf(acc[ot][r] + bsum);
      if (store_mode == 0) {
        out[(size_t)token * CI_ + o] = h;                    // token-major
      } else {
        const int b  = token >> 12;                          // / N_
        const int nl = token & (N_ - 1);
        out[((size_t)b * CI_ + o) * N_ + nl] = h;            // channel-major
      }
    }
  }
}

// ---------------------------------------------------------------------------
// 3) Fused flash attention, double-buffered async LDS staging of phi/G tiles.
//    Block = 4 waves = 64 queries; per 32-key step stage 24 KB shared by all
//    waves (4x L2-traffic reduction), overlap DMA of step i+1 with compute i.
// ---------------------------------------------------------------------------
__global__ void __launch_bounds__(128)
k_attn(const __bf16* __restrict__ Tt, const __bf16* __restrict__ Pm,
       const __bf16* __restrict__ G1, const __bf16* __restrict__ G2,
       __bf16* __restrict__ Y1, __bf16* __restrict__ Y2) {
  __shared__ __bf16 phiS[2][KB_ * CI_];            // [key][chan]   8 KB x2
  __shared__ __bf16 g1S[2][CI_ * KB_];             // [chan][key]   8 KB x2
  __shared__ __bf16 g2S[2][CI_ * KB_];             //               8 KB x2
  __shared__ __bf16 plds_all[4][16 * 32];          // P transpose, wave-private

  const int tid  = threadIdx.x;
  const int lane = tid & 31;
  const int wave = tid >> 5;
  __bf16* plds = plds_all[wave];
  const int g = lane >> 4, m = lane & 15;
  const int token0 = (blockIdx.x * 4 + wave) * 16;
  const int b = token0 >> 12;

  const __bf16* Pb  = Pm + (size_t)b * N_ * CI_;
  const __bf16* G1b = G1 + (size_t)b * CI_ * N_;
  const __bf16* G2b = G2 + (size_t)b * CI_ * N_;

  // Preload theta A-tiles for this query block (CI_=128 -> 4 k-steps of 32)
  v16bf At[4];
  {
    const __bf16* arow = Tt + (size_t)(token0 + m) * CI_;
#pragma unroll
    for (int kk = 0; kk < 4; ++kk) At[kk] = load_Atile(arow, kk * 32, g);
  }

  v8f acc1[8], acc2[8];
#pragma unroll
  for (int t = 0; t < 8; ++t) {
    acc1[t] = (v8f){0.f,0.f,0.f,0.f,0.f,0.f,0.f,0.f};
    acc2[t] = (v8f){0.f,0.f,0.f,0.f,0.f,0.f,0.f,0.f};
  }
  float rm[8], rl[8];
#pragma unroll
  for (int r = 0; r < 8; ++r) { rm[r] = -3.0e38f; rl[r] = 0.f; }

  // Stage one 32-key step into buffer bi. phi tile is 8 KB contiguous in
  // global (token-major); G rows are 64 B at 8 KB stride (channel-major).
  auto stage = [&](int bi, int kb) {
    {
      unsigned l0 = lds_off(&phiS[bi][0]) + (unsigned)tid * 64u;
      unsigned g0 = (unsigned)kb * (CI_ * 2) + (unsigned)tid * 64u;
#pragma unroll
      for (int i = 0; i < 4; ++i) async_b128(l0 + i * 16, g0 + i * 16, Pb);
    }
    {
      unsigned l1 = lds_off(&g1S[bi][0]) + (unsigned)tid * 64u;
      unsigned l2 = lds_off(&g2S[bi][0]) + (unsigned)tid * 64u;
      unsigned g0 = (unsigned)tid * (N_ * 2) + (unsigned)kb * 2u;
#pragma unroll
      for (int i = 0; i < 4; ++i) async_b128(l1 + i * 16, g0 + i * 16, G1b);
#pragma unroll
      for (int i = 0; i < 4; ++i) async_b128(l2 + i * 16, g0 + i * 16, G2b);
    }
  };

  stage(0, 0);
  for (int it = 0; it < N_ / KB_; ++it) {
    const int bi = it & 1;
    const int kb = it * KB_;
    wait_async0();                       // this wave's DMA for buffer bi done
    __syncthreads();                     // => everyone's portion visible
    if (it + 1 < N_ / KB_) stage(bi ^ 1, kb + KB_);

    // ---- S tiles: 16 queries x 32 keys, B from LDS -------------------------
    v8f s0 = (v8f){0.f,0.f,0.f,0.f,0.f,0.f,0.f,0.f};
    v8f s1 = (v8f){0.f,0.f,0.f,0.f,0.f,0.f,0.f,0.f};
#pragma unroll
    for (int kk = 0; kk < 4; ++kk) {
      v16bf B0 = *(const v16bf*)(&phiS[bi][(size_t)m * CI_ + kk * 32 + 16 * g]);
      v16bf B1 = *(const v16bf*)(&phiS[bi][(size_t)(16 + m) * CI_ + kk * 32 + 16 * g]);
      s0 = wmma_bf16(At[kk], B0, s0);
      s1 = wmma_bf16(At[kk], B1, s1);
    }
    // ---- online softmax across the 32 keys ---------------------------------
    float alpha[8];
#pragma unroll
    for (int r = 0; r < 8; ++r) {
      float mx = fmaxf(s0[r], s1[r]);                // row q = r + 8g
      mx = fmaxf(mx, __shfl_xor(mx, 1, 32));
      mx = fmaxf(mx, __shfl_xor(mx, 2, 32));
      mx = fmaxf(mx, __shfl_xor(mx, 4, 32));
      mx = fmaxf(mx, __shfl_xor(mx, 8, 32));
      const float nm = fmaxf(rm[r], mx);
      const float a  = __expf(rm[r] - nm);
      const float p0 = __expf(s0[r] - nm);
      const float p1 = __expf(s1[r] - nm);
      s0[r] = p0; s1[r] = p1;
      float ts = p0 + p1;
      ts += __shfl_xor(ts, 1, 32);
      ts += __shfl_xor(ts, 2, 32);
      ts += __shfl_xor(ts, 4, 32);
      ts += __shfl_xor(ts, 8, 32);
      rl[r] = rl[r] * a + ts;
      rm[r] = nm;
      alpha[r] = a;
    }
#pragma unroll
    for (int t = 0; t < 8; ++t) {
#pragma unroll
      for (int r = 0; r < 8; ++r) { acc1[t][r] *= alpha[r]; acc2[t][r] *= alpha[r]; }
    }
    // ---- transpose P (D layout -> A layout) through wave-private LDS -------
#pragma unroll
    for (int r = 0; r < 8; ++r) {
      const int q = r + 8 * g;
      plds[q * 32 + m]      = f2bf(s0[r]);
      plds[q * 32 + 16 + m] = f2bf(s1[r]);
    }
    asm volatile("s_wait_dscnt 0x0" ::: "memory");
    v16bf Pa = load_Atile_lds(plds + m * 32, g);
    // ---- y1/y2 accumulation: P(16x32) @ G(32keys x 128ch), B from LDS ------
#pragma unroll
    for (int ct = 0; ct < 8; ++ct) {
      const int c = ct * 16 + m;
      v16bf Bg1 = *(const v16bf*)(&g1S[bi][(size_t)c * KB_ + 16 * g]);
      acc1[ct] = wmma_bf16(Pa, Bg1, acc1[ct]);
      v16bf Bg2 = *(const v16bf*)(&g2S[bi][(size_t)c * KB_ + 16 * g]);
      acc2[ct] = wmma_bf16(Pa, Bg2, acc2[ct]);
    }
    __syncthreads();                     // all waves done reading buffer bi
  }
  // ---- normalize by row sums and store token-major bf16 --------------------
#pragma unroll
  for (int r = 0; r < 8; ++r) rl[r] = 1.0f / rl[r];
#pragma unroll
  for (int ct = 0; ct < 8; ++ct) {
    const int c = ct * 16 + m;
#pragma unroll
    for (int r = 0; r < 8; ++r) {
      const int token = token0 + r + 8 * g;
      Y1[(size_t)token * CI_ + c] = f2bf(acc1[ct][r] * rl[r]);
      Y2[(size_t)token * CI_ + c] = f2bf(acc2[ct][r] * rl[r]);
    }
  }
}

// ---------------------------------------------------------------------------
// 4) Output GEMM (CI->C) + fused BatchNorm(eval) + residual, f32 out (B,C,N)
//    Weight matrix staged into LDS via async DMA.
// ---------------------------------------------------------------------------
__global__ void __launch_bounds__(128)
k_out(const __bf16* __restrict__ Y, const __bf16* __restrict__ Wb,
      const float* __restrict__ wbias, const float* __restrict__ bng,
      const float* __restrict__ bnb, const float* __restrict__ bnm,
      const float* __restrict__ bnv, const float* __restrict__ fea,
      float* __restrict__ Z) {
  __shared__ __bf16 wS[C_ * CI_];                  // 64 KB
  const int tid  = threadIdx.x;
  const int lane = tid & 31;
  const int wave = tid >> 5;
  const int g = lane >> 4, m = lane & 15;
  const int token0 = (blockIdx.x * 4 + wave) * 16;

  {
    unsigned l0 = lds_off(wS) + (unsigned)tid * 512u;
    unsigned g0 = (unsigned)tid * 512u;
#pragma unroll
    for (int i = 0; i < 32; ++i)
      async_b128(l0 + i * 16, g0 + i * 16, Wb);
  }
  wait_async0();
  __syncthreads();

  v8f acc[16];
#pragma unroll
  for (int t = 0; t < 16; ++t) acc[t] = (v8f){0.f,0.f,0.f,0.f,0.f,0.f,0.f,0.f};

  const __bf16* arow = Y + (size_t)(token0 + m) * CI_;
#pragma unroll
  for (int kk = 0; kk < CI_; kk += 32) {
    v16bf A = load_Atile(arow, kk, g);
#pragma unroll
    for (int ot = 0; ot < 16; ++ot) {
      v16bf Bt = *(const v16bf*)(wS + (size_t)(ot * 16 + m) * CI_ + kk + 16 * g);
      acc[ot] = wmma_bf16(A, Bt, acc[ot]);
    }
  }
  const int b  = token0 >> 12;
  const int nl = token0 & (N_ - 1);
#pragma unroll
  for (int ot = 0; ot < 16; ++ot) {
    const int o = ot * 16 + m;
    const float scale = bng[o] * rsqrtf(bnv[o] + EPSV);
    const float shift = (wbias[o] - bnm[o]) * scale + bnb[o];
    const float* frow = fea + ((size_t)b * C_ + o) * N_ + nl;
    float*       zrow = Z   + ((size_t)b * C_ + o) * N_ + nl;
#pragma unroll
    for (int r = 0; r < 8; ++r) {
      const int n = r + 8 * g;
      zrow[n] = acc[ot][r] * scale + shift + frow[n];
    }
  }
}

// ---------------------------------------------------------------------------
extern "C" void kernel_launch(void* const* d_in, const int* in_sizes, int n_in,
                              void* d_out, int out_size, void* d_ws, size_t ws_size,
                              hipStream_t stream) {
  const float* fea1    = (const float*)d_in[0];
  const float* fea2    = (const float*)d_in[1];
  const float* g1_w    = (const float*)d_in[2];
  const float* g1_b    = (const float*)d_in[3];
  const float* g2_w    = (const float*)d_in[4];
  const float* g2_b    = (const float*)d_in[5];
  const float* theta_w = (const float*)d_in[6];
  const float* theta_b = (const float*)d_in[7];
  const float* phi_w   = (const float*)d_in[8];
  const float* phi_b   = (const float*)d_in[9];
  const float* w1_w    = (const float*)d_in[10];
  const float* w1_b    = (const float*)d_in[11];
  const float* bn1_g   = (const float*)d_in[12];
  const float* bn1_b   = (const float*)d_in[13];
  const float* bn1_m   = (const float*)d_in[14];
  const float* bn1_v   = (const float*)d_in[15];
  const float* w2_w    = (const float*)d_in[16];
  const float* w2_b    = (const float*)d_in[17];
  const float* bn2_g   = (const float*)d_in[18];
  const float* bn2_b   = (const float*)d_in[19];
  const float* bn2_m   = (const float*)d_in[20];
  const float* bn2_v   = (const float*)d_in[21];

  char* ws = (char*)d_ws;
  size_t off = 0;
  auto alloc = [&](size_t nbytes) {
    char* p = ws + off;
    off += (nbytes + 255) & ~(size_t)255;
    return p;
  };
  const size_t tokBytesC  = (size_t)Bn * N_ * C_  * 2;   // 8 MB
  const size_t tokBytesCI = (size_t)Bn * N_ * CI_ * 2;   // 4 MB
  const size_t wBytes     = (size_t)CI_ * C_ * 2;        // 64 KB

  __bf16* XT1 = (__bf16*)alloc(tokBytesC);
  __bf16* XT2 = (__bf16*)alloc(tokBytesC);
  __bf16* WTH = (__bf16*)alloc(wBytes);
  __bf16* WPH = (__bf16*)alloc(wBytes);
  __bf16* WG1 = (__bf16*)alloc(wBytes);
  __bf16* WG2 = (__bf16*)alloc(wBytes);
  __bf16* W1B = (__bf16*)alloc(wBytes);
  __bf16* W2B = (__bf16*)alloc(wBytes);
  __bf16* Tt  = (__bf16*)alloc(tokBytesCI);
  __bf16* Pt  = (__bf16*)alloc(tokBytesCI);
  __bf16* GT1 = (__bf16*)alloc(tokBytesCI);
  __bf16* GT2 = (__bf16*)alloc(tokBytesCI);
  __bf16* Y1  = (__bf16*)alloc(tokBytesCI);
  __bf16* Y2  = (__bf16*)alloc(tokBytesCI);
  (void)ws_size; (void)n_in; (void)in_sizes; (void)out_size;

  // 1) pack/convert
  dim3 tg(N_ / 32, C_ / 32, Bn);
  k_pack_tokens<<<tg, dim3(32, 8), 0, stream>>>(fea1, XT1);
  k_pack_tokens<<<tg, dim3(32, 8), 0, stream>>>(fea2, XT2);
  const int wn = CI_ * C_;
  k_cvt<<<(wn + 255) / 256, 256, 0, stream>>>(theta_w, WTH, wn);
  k_cvt<<<(wn + 255) / 256, 256, 0, stream>>>(phi_w,   WPH, wn);
  k_cvt<<<(wn + 255) / 256, 256, 0, stream>>>(g1_w,    WG1, wn);
  k_cvt<<<(wn + 255) / 256, 256, 0, stream>>>(g2_w,    WG2, wn);
  k_cvt<<<(wn + 255) / 256, 256, 0, stream>>>(w1_w,    W1B, wn);
  k_cvt<<<(wn + 255) / 256, 256, 0, stream>>>(w2_w,    W2B, wn);

  // 2) projections (grid: Bn*N/(16 tokens * 4 waves) = 256 blocks)
  const int pg = Bn * N_ / 64;
  k_proj<<<pg, 128, 0, stream>>>(XT1, WTH, theta_b, Tt,  0);
  k_proj<<<pg, 128, 0, stream>>>(XT2, WPH, phi_b,   Pt,  0);
  k_proj<<<pg, 128, 0, stream>>>(XT1, WG1, g1_b,    GT1, 1);
  k_proj<<<pg, 128, 0, stream>>>(XT2, WG2, g2_b,    GT2, 1);

  // 3) fused flash attention (double-buffered async LDS staging)
  k_attn<<<pg, 128, 0, stream>>>(Tt, Pt, GT1, GT2, Y1, Y2);

  // 4) output GEMM + BN + residual
  float* Z1 = (float*)d_out;
  float* Z2 = (float*)d_out + (size_t)Bn * C_ * N_;
  k_out<<<pg, 128, 0, stream>>>(Y1, W1B, w1_b, bn1_g, bn1_b, bn1_m, bn1_v, fea1, Z1);
  k_out<<<pg, 128, 0, stream>>>(Y2, W2B, w2_b, bn2_g, bn2_b, bn2_m, bn2_v, fea2, Z2);
}
